// GCN_26525718020488
// MI455X (gfx1250) — compile-verified, matrix operand order checked by reading, not run on hardware
//
#include <hip/hip_runtime.h>
#include <hip/hip_bf16.h>

// ---------------------------------------------------------------------------
// 2-layer GCN (DGL GraphConv, norm='both') for MI455X (gfx1250, wave32).
//   h1 = relu( Ddst^-1/2 * A * Dsrc^-1/2 * (X W1) + b1 )
//   out =       Ddst^-1/2 * A * Dsrc^-1/2 * (h1 W2) + b2
// Dense GEMMs: v_wmma_f32_16x16x32_bf16, fully unrolled over K (templated),
//   A panel staged in LDS (one cooperative load per block), weights pre-packed
//   to bf16 col-major so B fragments are contiguous b128 loads.
// Sparse aggregation: L2-resident float4 gathers + global_atomic_add_f32
//   (feature matrix is 51 MB < 192 MB L2, so edge gathers/atomics stay on-chip).
// ---------------------------------------------------------------------------

#define GCN_N_NODES  100000
#define GCN_N_EDGES  1600000
#define GCN_IN_F     128
#define GCN_HID      128
#define GCN_NCLS     64

typedef __attribute__((ext_vector_type(16))) __bf16 v16bf;
typedef __attribute__((ext_vector_type(8)))  __bf16 v8bf;
typedef __attribute__((ext_vector_type(8)))  float  v8f;

// ---------------------------------------------------------------- utilities

__global__ void gcn_zero_kernel(float* __restrict__ p, long n) {
    long i = (long)blockIdx.x * blockDim.x + threadIdx.x;
    if (i < n) p[i] = 0.0f;
}

__global__ void gcn_degree_kernel(const int* __restrict__ src,
                                  const int* __restrict__ dst,
                                  float* __restrict__ deg_src,
                                  float* __restrict__ deg_dst, int E) {
    int e = blockIdx.x * blockDim.x + threadIdx.x;
    if (e >= E) return;
    atomicAdd(deg_src + src[e], 1.0f);
    atomicAdd(deg_dst + dst[e], 1.0f);
}

// in-place: deg -> (deg > 0 ? deg^-1/2 : 0)
__global__ void gcn_norm_kernel(float* __restrict__ p, int n) {
    int i = blockIdx.x * blockDim.x + threadIdx.x;
    if (i >= n) return;
    float d = p[i];
    p[i] = (d > 0.0f) ? rsqrtf(d) : 0.0f;
}

// Pack W[k, c] (KxNC row-major fp32) -> Wt[c, k] (col-major bf16).
// Makes each lane's B fragment two contiguous 16B loads in the GEMM.
__global__ void gcn_pack_wt_bf16(const float* __restrict__ W,
                                 __bf16* __restrict__ Wt, int K, int NC) {
    int i = blockIdx.x * blockDim.x + threadIdx.x;
    if (i >= K * NC) return;
    int c = i / K;
    int k = i - c * K;
    Wt[i] = (__bf16)W[(size_t)k * NC + c];
}

// ------------------------------------------------------------- WMMA GEMM
// out[r, c] = norm[r] * sum_k X[r,k] * Wt[c,k]
// One wave computes one 16x16 tile; 4 waves/block share a 16xK LDS A-panel.

template <int K, int NC>
__global__ __launch_bounds__(128)
void gcn_gemm_wmma_bf16(const float* __restrict__ X,
                        const __bf16* __restrict__ Wt,   // [NC][K] packed bf16
                        const float* __restrict__ normRow,
                        float* __restrict__ out) {
    __shared__ float sA[16 * K];                  // 8 KB for K=128

    const int tid     = threadIdx.x;
    const int lane    = tid & 31;
    const int wave    = tid >> 5;
    const int rowTile = blockIdx.x;               // 16 rows
    const int colTile = blockIdx.y * 4 + wave;    // 16 cols
    const int khalf   = lane >> 4;                // K-half select (ISA A/B layout)
    const int col     = colTile * 16 + (lane & 15);
    const int row0    = rowTile * 16;

    // Cooperative A-panel load: 16 rows x K fp32, float4-vectorized.
    constexpr int F4 = (16 * K) / 4;
    const float4* gx = (const float4*)(X + (size_t)row0 * K);
#pragma unroll
    for (int i = tid; i < F4; i += 128) {
        ((float4*)sA)[i] = gx[i];
    }
    __syncthreads();

    v8f c = {};
#pragma unroll
    for (int kb = 0; kb < K; kb += 32) {
        // ---- A fragment from LDS: lane holds K = kb + khalf*8 + {0..7, 16..23}.
        const float* ap = sA + (lane & 15) * K + kb + khalf * 8;
        float4 a0 = *(const float4*)(ap +  0);
        float4 a1 = *(const float4*)(ap +  4);
        float4 a2 = *(const float4*)(ap + 16);
        float4 a3 = *(const float4*)(ap + 20);
        v16bf a;
        a[0]  = (__bf16)a0.x; a[1]  = (__bf16)a0.y; a[2]  = (__bf16)a0.z; a[3]  = (__bf16)a0.w;
        a[4]  = (__bf16)a1.x; a[5]  = (__bf16)a1.y; a[6]  = (__bf16)a1.z; a[7]  = (__bf16)a1.w;
        a[8]  = (__bf16)a2.x; a[9]  = (__bf16)a2.y; a[10] = (__bf16)a2.z; a[11] = (__bf16)a2.w;
        a[12] = (__bf16)a3.x; a[13] = (__bf16)a3.y; a[14] = (__bf16)a3.z; a[15] = (__bf16)a3.w;

        // ---- B fragment: contiguous bf16, two 16B loads (same K striping).
        const __bf16* bp = Wt + (size_t)col * K + kb + khalf * 8;
        v8bf b0 = *(const v8bf*)(bp +  0);
        v8bf b1 = *(const v8bf*)(bp + 16);
        v16bf b = __builtin_shufflevector(b0, b1,
                                          0, 1, 2, 3, 4, 5, 6, 7,
                                          8, 9, 10, 11, 12, 13, 14, 15);

        // D = A x B + C  (fp32 accumulate)
        c = __builtin_amdgcn_wmma_f32_16x16x32_bf16(
                /*neg_a=*/false, a, /*neg_b=*/false, b,
                /*c_mod=*/(short)0, c, /*reuse_a=*/false, /*reuse_b=*/false);
    }

    // C/D layout: VGPR i -> M = i + 8*khalf, N = lane & 15. Fuse norm_src scale.
    const int rbase = row0 + khalf * 8;
#pragma unroll
    for (int i = 0; i < 8; ++i) {
        int r = rbase + i;
        out[(size_t)r * NC + col] = c[i] * normRow[r];
    }
}

// ------------------------------------------------------ sparse aggregation
// m[dst[e], :] += h[src[e], :], float4-vectorized; one thread per (edge, 4 feats).
__global__ void gcn_aggregate_kernel(const float* __restrict__ h,
                                     const int* __restrict__ src,
                                     const int* __restrict__ dst,
                                     float* __restrict__ m,
                                     int E, int f4shift, int F) {
    long idx = (long)blockIdx.x * blockDim.x + threadIdx.x;
    long total = (long)E << f4shift;
    if (idx >= total) return;
    int e = (int)(idx >> f4shift);
    int f = (int)(idx & ((1 << f4shift) - 1)) * 4;
    int s = src[e];
    int d = dst[e];
    float4 v = *(const float4*)(h + (size_t)s * F + f);
    float* mp = m + (size_t)d * F + f;
    atomicAdd(mp + 0, v.x);
    atomicAdd(mp + 1, v.y);
    atomicAdd(mp + 2, v.z);
    atomicAdd(mp + 3, v.w);
}

// out = (relu?) ( m * norm_dst[row] + bias[col] )
__global__ void gcn_finalize_kernel(const float* __restrict__ m,
                                    const float* __restrict__ norm_dst,
                                    const float* __restrict__ bias,
                                    float* __restrict__ out,
                                    int total, int fshift, int fmask, int do_relu) {
    int idx = blockIdx.x * blockDim.x + threadIdx.x;
    if (idx >= total) return;
    int r = idx >> fshift;
    int cc = idx & fmask;
    float v = m[idx] * norm_dst[r] + bias[cc];
    out[idx] = do_relu ? fmaxf(v, 0.0f) : v;
}

// ---------------------------------------------------------------- launcher

extern "C" void kernel_launch(void* const* d_in, const int* in_sizes, int n_in,
                              void* d_out, int out_size, void* d_ws, size_t ws_size,
                              hipStream_t stream) {
    (void)in_sizes; (void)n_in; (void)out_size; (void)ws_size;

    const float* x   = (const float*)d_in[0];
    const int*   src = (const int*)  d_in[1];
    const int*   dst = (const int*)  d_in[2];
    const float* W1  = (const float*)d_in[3];
    const float* b1  = (const float*)d_in[4];
    const float* W2  = (const float*)d_in[5];
    const float* b2  = (const float*)d_in[6];
    float*       out = (float*)d_out;

    const int N = GCN_N_NODES;
    const int E = GCN_N_EDGES;

    // Workspace (floats): [norm_src N][norm_dst N][bufA N*128][bufB N*128][Wt1][Wt2]
    float* ws       = (float*)d_ws;
    float* norm_src = ws;                                   // N
    float* norm_dst = ws + N;                               // N
    float* bufA     = ws + 2 * (size_t)N;                   // N*128
    float* bufB     = bufA + (size_t)N * GCN_HID;           // N*128
    __bf16* wt1     = (__bf16*)(bufB + (size_t)N * GCN_HID);        // 128*128 bf16
    __bf16* wt2     = wt1 + (size_t)GCN_IN_F * GCN_HID;             // 128*64 bf16

    const int TB = 256;

    // 0) pack weights to bf16 col-major (tiny, once per call)
    gcn_pack_wt_bf16<<<(GCN_IN_F * GCN_HID + TB - 1) / TB, TB, 0, stream>>>(
        W1, wt1, GCN_IN_F, GCN_HID);
    gcn_pack_wt_bf16<<<(GCN_HID * GCN_NCLS + TB - 1) / TB, TB, 0, stream>>>(
        W2, wt2, GCN_HID, GCN_NCLS);

    // 1) degrees -> norms (both arrays contiguous: one zero + one norm pass)
    gcn_zero_kernel<<<(2 * N + TB - 1) / TB, TB, 0, stream>>>(norm_src, 2L * N);
    gcn_degree_kernel<<<(E + TB - 1) / TB, TB, 0, stream>>>(src, dst, norm_src, norm_dst, E);
    gcn_norm_kernel<<<(2 * N + TB - 1) / TB, TB, 0, stream>>>(norm_src, 2 * N);

    // 2) layer 1 GEMM: bufA = (X @ W1) * norm_src[:,None]   (100000x128)
    gcn_gemm_wmma_bf16<GCN_IN_F, GCN_HID>
        <<<dim3(N / 16, GCN_HID / 64), 128, 0, stream>>>(x, wt1, norm_src, bufA);

    // 3) layer 1 aggregation: bufB = segment_sum(bufA[src], dst)
    gcn_zero_kernel<<<((long)N * GCN_HID + TB - 1) / TB, TB, 0, stream>>>(
        bufB, (long)N * GCN_HID);
    gcn_aggregate_kernel<<<(((long)E << 5) + TB - 1) / TB, TB, 0, stream>>>(
        bufA, src, dst, bufB, E, /*f4shift=*/5, GCN_HID);

    // 4) layer 1 finalize: bufA = relu(bufB * norm_dst + b1)
    gcn_finalize_kernel<<<(N * GCN_HID + TB - 1) / TB, TB, 0, stream>>>(
        bufB, norm_dst, b1, bufA, N * GCN_HID, /*fshift=*/7, /*fmask=*/127, /*relu=*/1);

    // 5) layer 2 GEMM: bufB[:, :64] = (bufA @ W2) * norm_src[:,None]  (100000x64)
    gcn_gemm_wmma_bf16<GCN_HID, GCN_NCLS>
        <<<dim3(N / 16, GCN_NCLS / 64), 128, 0, stream>>>(bufA, wt2, norm_src, bufB);

    // 6) layer 2 aggregation into bufA[:, :64] (GEMM2 finished reading bufA)
    gcn_zero_kernel<<<((long)N * GCN_NCLS + TB - 1) / TB, TB, 0, stream>>>(
        bufA, (long)N * GCN_NCLS);
    gcn_aggregate_kernel<<<(((long)E << 4) + TB - 1) / TB, TB, 0, stream>>>(
        bufB, src, dst, bufA, E, /*f4shift=*/4, GCN_NCLS);

    // 7) layer 2 finalize: out = bufA * norm_dst + b2
    gcn_finalize_kernel<<<(N * GCN_NCLS + TB - 1) / TB, TB, 0, stream>>>(
        bufA, norm_dst, b2, out, N * GCN_NCLS, /*fshift=*/6, /*fmask=*/63, /*relu=*/0);
}
